// SelfAttentionLayer_498216206636
// MI455X (gfx1250) — matrix operand order repaired and checked by reading, not compiled, not run on hardware
//
#include <hip/hip_runtime.h>
#include <hip/hip_bf16.h>

typedef __attribute__((ext_vector_type(16))) _Float16 v16h;
typedef __attribute__((ext_vector_type(8)))  _Float16 v8h;
typedef __attribute__((ext_vector_type(4)))  _Float16 v4h;
typedef __attribute__((ext_vector_type(8)))  float    v8f;
typedef __attribute__((ext_vector_type(4)))  float    v4f;
typedef __attribute__((ext_vector_type(4)))  int      v4i;

#define N_HEADS 16
#define EMBED   1024
#define HD      64
#define BATCH   4
#define SEQ     2048
#define M_TOT   (BATCH * SEQ)   // 8192
#define TPITCH  40              // staged-tile row pitch in halves (80 B, 16B-aligned)

// ---------------------------------------------------------------------------
// WMMA helper: D = A(16x32 f16) * B(32x16 f16) + C(16x16 f32)
// ---------------------------------------------------------------------------
__device__ __forceinline__ v8f wmma_f16(v16h a, v16h b, v8f c) {
    return __builtin_amdgcn_wmma_f32_16x16x32_f16(
        false, a, false, b, (short)0, c, false, false);
}

// ---------------------------------------------------------------------------
// Async global->LDS copy of 16 bytes (GLOBAL_LOAD_ASYNC_TO_LDS_B128, ASYNCcnt)
// builtin signature wants v4i pointers: (as1 v4i* src, as3 v4i* dst, imm, cpol)
// ---------------------------------------------------------------------------
typedef __attribute__((address_space(1))) v4i GV4;
typedef __attribute__((address_space(3))) v4i LV4;

__device__ __forceinline__ void async_b128(const _Float16* g, _Float16* l) {
#if __has_builtin(__builtin_amdgcn_global_load_async_to_lds_b128)
    __builtin_amdgcn_global_load_async_to_lds_b128((GV4*)g, (LV4*)l, 0, 0);
#else
    unsigned loff = (unsigned)(size_t)(__attribute__((address_space(3))) _Float16*)l;
    unsigned long long ga = (unsigned long long)g;
    asm volatile("global_load_async_to_lds_b128 %0, %1, off"
                 :: "v"(loff), "v"(ga) : "memory");
#endif
}

__device__ __forceinline__ void wait_async0() {
    asm volatile("s_wait_asynccnt 0x0" ::: "memory");
}

// A-fragment (16x32, rows along M): lanes 0-15 hold K {0..7,16..23},
// lanes 16-31 hold K {8..15,24..31}; row = lane%16, K contiguous in source.
__device__ __forceinline__ v16h frag_a(const _Float16* __restrict__ p) {
    v8h lo = *(const v8h*)p;
    v8h hi = *(const v8h*)(p + 16);
    v16h r;
#pragma unroll
    for (int i = 0; i < 8; ++i) { r[i] = lo[i]; r[8 + i] = hi[i]; }
    return r;
}
// B-fragment (32x16, cols along N): lane holds column n = lane%16; lanes 0-15
// hold K {0..7,16..23}, lanes 16-31 K {8..15,24..31}.  Source row-major along the column's K.
__device__ __forceinline__ v16h frag_b(const _Float16* __restrict__ p) {
    v8h lo = *(const v8h*)p;
    v8h hi = *(const v8h*)(p + 16);
    v16h r;
#pragma unroll
    for (int i = 0; i < 8; ++i) { r[i] = lo[i]; r[8 + i] = hi[i]; }
    return r;
}
__device__ __forceinline__ v16h load_frag_a(const _Float16* __restrict__ base,
                                            int ld, int row, int k0, int lane) {
    return frag_a(base + (size_t)row * ld + k0 + ((lane & 16) ? 8 : 0));
}
__device__ __forceinline__ v16h load_frag_b(const _Float16* __restrict__ base,
                                            int ld, int col, int k0, int lane) {
    return frag_b(base + (size_t)col * ld + k0 + ((lane & 16) ? 8 : 0));
}

// ---------------------------------------------------------------------------
// fp32 -> f16 conversion (4 elements / thread, exact grids)
// ---------------------------------------------------------------------------
__global__ void __launch_bounds__(256) cvt_f32_f16(const float* __restrict__ src,
                                                   _Float16* __restrict__ dst) {
    int idx = (blockIdx.x * 256 + threadIdx.x) * 4;
    v4f v = *(const v4f*)(src + idx);
    v4h o;
#pragma unroll
    for (int i = 0; i < 4; ++i) o[i] = (_Float16)v[i];
    *(v4h*)(dst + idx) = o;
}

// ---------------------------------------------------------------------------
// QKV projection: y = x @ W^T, LDS-tiled.  128x128 block tile, k-steps of 32
// double-buffered via async global->LDS copies.  Each wave (2x4 wave grid)
// computes a 64x32 sub-tile = 4x2 WMMA accumulators with fragment reuse.
// blockIdx.y: 0=Q, 1=K (stored [bh][s][64]); 2=V stored transposed [bh][64][s].
// ---------------------------------------------------------------------------
__global__ void __launch_bounds__(256) qkv_gemm(const _Float16* __restrict__ xh,
                                                const _Float16* __restrict__ wq,
                                                const _Float16* __restrict__ wk,
                                                const _Float16* __restrict__ wv,
                                                _Float16* __restrict__ qh,
                                                _Float16* __restrict__ kh,
                                                _Float16* __restrict__ vth) {
    __shared__ _Float16 smem[2][2][128 * TPITCH];   // [buf][x|w][row*TPITCH+k]

    const int tid  = threadIdx.x;
    const int lane = tid & 31, wave = tid >> 5;
    const int l15 = lane & 15, hilo = (lane >> 4) & 1;
    const int bm = blockIdx.x >> 3;                 // 0..63  (128-row block)
    const int bn = blockIdx.x & 7;                  // 0..7   (128-col block)
    const int z  = blockIdx.y;                      // 0,1,2
    const _Float16* __restrict__ W = (z == 0) ? wq : (z == 1) ? wk : wv;
    const int wm = wave >> 2, wn = wave & 3;        // 2x4 wave grid

    // async staging map: 256 threads x 2 b128 per tile (x and W)
    const int srow = tid >> 1;                      // 0..127
    const int scol = (tid & 1) * 16;                // halves: 0 or 16
    const _Float16* gx = xh + (size_t)(bm * 128 + srow) * EMBED + scol;
    const _Float16* gw = W  + (size_t)(bn * 128 + srow) * EMBED + scol;

    v8f acc[4][2] = {};
    int buf = 0;

    // prologue: stage k-chunk 0 into buffer 0
    {
        _Float16* dx = &smem[0][0][srow * TPITCH + scol];
        _Float16* dw = &smem[0][1][srow * TPITCH + scol];
        async_b128(gx, dx);     async_b128(gx + 8, dx + 8);
        async_b128(gw, dw);     async_b128(gw + 8, dw + 8);
    }

    for (int ks = 0; ks < EMBED / 32; ++ks) {
        wait_async0();          // own wave's async copies into `buf` landed
        __syncthreads();        // everyone's copies landed, prev compute done
        if (ks + 1 < EMBED / 32) {  // refill the other buffer (read-free now)
            const int k0 = (ks + 1) * 32;
            _Float16* dx = &smem[buf ^ 1][0][srow * TPITCH + scol];
            _Float16* dw = &smem[buf ^ 1][1][srow * TPITCH + scol];
            async_b128(gx + k0, dx);     async_b128(gx + k0 + 8, dx + 8);
            async_b128(gw + k0, dw);     async_b128(gw + k0 + 8, dw + 8);
        }
        const _Float16* xs = &smem[buf][0][0];
        const _Float16* wsm = &smem[buf][1][0];
        v16h a[4], b[2];
#pragma unroll
        for (int i = 0; i < 4; ++i)
            a[i] = frag_a(xs + (wm * 64 + i * 16 + l15) * TPITCH + hilo * 8);
#pragma unroll
        for (int j = 0; j < 2; ++j)
            b[j] = frag_b(wsm + (wn * 32 + j * 16 + l15) * TPITCH + hilo * 8);
#pragma unroll
        for (int i = 0; i < 4; ++i)
#pragma unroll
            for (int j = 0; j < 2; ++j)
                acc[i][j] = wmma_f16(a[i], b[j], acc[i][j]);
        buf ^= 1;
    }

    // epilogue: 8 16x16 tiles per wave
#pragma unroll
    for (int i = 0; i < 4; ++i) {
        const int r0 = bm * 128 + wm * 64 + i * 16 + hilo * 8;
        const int b = r0 >> 11;          // batch
        const int s0 = r0 & 2047;        // sequence
#pragma unroll
        for (int j = 0; j < 2; ++j) {
            const int n = bn * 128 + wn * 32 + j * 16 + l15;
            const int h = n >> 6, d = n & 63;
            if (z == 2) {
                v8h vv;
#pragma unroll
                for (int e = 0; e < 8; ++e) vv[e] = (_Float16)acc[i][j][e];
                *(v8h*)(vth + ((size_t)(b * N_HEADS + h) * HD + d) * SEQ + s0) = vv;
            } else {
                _Float16* dst = (z == 0 ? qh : kh) +
                                ((size_t)(b * N_HEADS + h) * SEQ + s0) * HD + d;
#pragma unroll
                for (int e = 0; e < 8; ++e) dst[(size_t)e * HD] = (_Float16)acc[i][j][e];
            }
        }
    }
}

// ---------------------------------------------------------------------------
// Flash attention: one wave owns 16 queries of one (b,h); single pass over all
// 2048 keys in chunks of 32 with online softmax.  scale = 1/sqrt(EMBED) = 1/32.
// ---------------------------------------------------------------------------
__global__ void __launch_bounds__(256) attn_kernel(const _Float16* __restrict__ qh,
                                                   const _Float16* __restrict__ kh,
                                                   const _Float16* __restrict__ vth,
                                                   float* __restrict__ out) {
    __shared__ _Float16 pbuf[8][16][48];       // per-wave 16x32 P tile, 96B pitch

    const int lane = threadIdx.x & 31;
    const int wave = threadIdx.x >> 5;
    const int gid = blockIdx.x * 8 + wave;     // 0..8191
    const int bh = gid >> 7;                   // b*16+h
    const int qt = gid & 127;                  // query tile
    const int hilo = (lane >> 4) & 1;
    const int l15 = lane & 15;

    const _Float16* __restrict__ Q  = qh  + (size_t)bh * SEQ * HD;
    const _Float16* __restrict__ K  = kh  + (size_t)bh * SEQ * HD;
    const _Float16* __restrict__ Vt = vth + (size_t)bh * HD * SEQ;

    const v16h aq0 = load_frag_a(Q, HD, qt * 16 + l15, 0, lane);
    const v16h aq1 = load_frag_a(Q, HD, qt * 16 + l15, 32, lane);

    float mrow[8], lrow[8];
    v8f o0 = {}, o1 = {}, o2 = {}, o3 = {};
#pragma unroll
    for (int i = 0; i < 8; ++i) { mrow[i] = -3.0e38f; lrow[i] = 0.0f; }
    const float SCALE = 0.03125f;              // 1/sqrt(1024)

    for (int kc = 0; kc < SEQ / 32; ++kc) {
        const int kb = kc * 32;
        if (kc + 1 < SEQ / 32)
            __builtin_prefetch(K + (size_t)(kb + 32 + l15) * HD, 0, 0);

        v16h bk0a = load_frag_b(K, HD, kb + l15, 0, lane);
        v16h bk0b = load_frag_b(K, HD, kb + l15, 32, lane);
        v16h bk1a = load_frag_b(K, HD, kb + 16 + l15, 0, lane);
        v16h bk1b = load_frag_b(K, HD, kb + 16 + l15, 32, lane);
        v8f s_0 = {}, s_1 = {};
        s_0 = wmma_f16(aq0, bk0a, s_0);
        s_0 = wmma_f16(aq1, bk0b, s_0);
        s_1 = wmma_f16(aq0, bk1a, s_1);
        s_1 = wmma_f16(aq1, bk1b, s_1);

        float p0[8], p1[8], alpha[8];
#pragma unroll
        for (int i = 0; i < 8; ++i) {
            float a0 = s_0[i] * SCALE, a1 = s_1[i] * SCALE;
            float cm = fmaxf(a0, a1);
            cm = fmaxf(cm, __shfl_xor(cm, 1, 32));
            cm = fmaxf(cm, __shfl_xor(cm, 2, 32));
            cm = fmaxf(cm, __shfl_xor(cm, 4, 32));
            cm = fmaxf(cm, __shfl_xor(cm, 8, 32));
            float mn = fmaxf(mrow[i], cm);
            alpha[i] = __expf(mrow[i] - mn);
            mrow[i] = mn;
            p0[i] = __expf(a0 - mn);
            p1[i] = __expf(a1 - mn);
            float rs = p0[i] + p1[i];
            rs += __shfl_xor(rs, 1, 32);
            rs += __shfl_xor(rs, 2, 32);
            rs += __shfl_xor(rs, 4, 32);
            rs += __shfl_xor(rs, 8, 32);
            lrow[i] = lrow[i] * alpha[i] + rs;
        }

#pragma unroll
        for (int i = 0; i < 8; ++i) {
            o0[i] *= alpha[i]; o1[i] *= alpha[i];
            o2[i] *= alpha[i]; o3[i] *= alpha[i];
            pbuf[wave][hilo * 8 + i][l15]      = (_Float16)p0[i];
            pbuf[wave][hilo * 8 + i][16 + l15] = (_Float16)p1[i];
        }

        // re-read P in A-fragment layout; per-wave LDS region, DS ops in-order
        const _Float16* pp = &pbuf[wave][l15][hilo * 8];
        v8h plo = *(const v8h*)pp;
        v8h phi = *(const v8h*)(pp + 16);
        v16h ap;
#pragma unroll
        for (int i = 0; i < 8; ++i) { ap[i] = plo[i]; ap[8 + i] = phi[i]; }

        v16h bv0 = load_frag_b(Vt, SEQ,  0 + l15, kb, lane);
        v16h bv1 = load_frag_b(Vt, SEQ, 16 + l15, kb, lane);
        v16h bv2 = load_frag_b(Vt, SEQ, 32 + l15, kb, lane);
        v16h bv3 = load_frag_b(Vt, SEQ, 48 + l15, kb, lane);
        o0 = wmma_f16(ap, bv0, o0);
        o1 = wmma_f16(ap, bv1, o1);
        o2 = wmma_f16(ap, bv2, o2);
        o3 = wmma_f16(ap, bv3, o3);
    }

    const int b = bh >> 4, h = bh & 15;
    float* dst = out + ((size_t)b * SEQ + qt * 16 + hilo * 8) * EMBED + h * HD;
#pragma unroll
    for (int i = 0; i < 8; ++i) {
        float inv = 1.0f / lrow[i];
        dst[(size_t)i * EMBED +  0 + l15] = o0[i] * inv;
        dst[(size_t)i * EMBED + 16 + l15] = o1[i] * inv;
        dst[(size_t)i * EMBED + 32 + l15] = o2[i] * inv;
        dst[(size_t)i * EMBED + 48 + l15] = o3[i] * inv;
    }
}

// ---------------------------------------------------------------------------
extern "C" void kernel_launch(void* const* d_in, const int* in_sizes, int n_in,
                              void* d_out, int out_size, void* d_ws, size_t ws_size,
                              hipStream_t stream) {
    const float* x  = (const float*)d_in[0];
    const float* Wq = (const float*)d_in[1];
    const float* Wk = (const float*)d_in[2];
    const float* Wv = (const float*)d_in[3];
    float* out = (float*)d_out;

    _Float16* ws = (_Float16*)d_ws;
    size_t off = 0;
    _Float16* xh  = ws + off; off += (size_t)M_TOT * EMBED;
    _Float16* wqh = ws + off; off += (size_t)EMBED * EMBED;
    _Float16* wkh = ws + off; off += (size_t)EMBED * EMBED;
    _Float16* wvh = ws + off; off += (size_t)EMBED * EMBED;
    _Float16* qh  = ws + off; off += (size_t)M_TOT * EMBED;   // [bh][s][64]
    _Float16* kh  = ws + off; off += (size_t)M_TOT * EMBED;   // [bh][s][64]
    _Float16* vth = ws + off; off += (size_t)M_TOT * EMBED;   // [bh][64][s]

    cvt_f32_f16<<<(M_TOT * EMBED) / (256 * 4), 256, 0, stream>>>(x, xh);
    cvt_f32_f16<<<(EMBED * EMBED) / (256 * 4), 256, 0, stream>>>(Wq, wqh);
    cvt_f32_f16<<<(EMBED * EMBED) / (256 * 4), 256, 0, stream>>>(Wk, wkh);
    cvt_f32_f16<<<(EMBED * EMBED) / (256 * 4), 256, 0, stream>>>(Wv, wvh);

    // 64 x 8 block tiles of 128x128, 3 matrices
    qkv_gemm<<<dim3(512, 3), 256, 0, stream>>>(xh, wqh, wkh, wvh, qh, kh, vth);

    // 64 (b,h) x 128 query tiles = 8192 waves -> 1024 blocks
    attn_kernel<<<1024, 256, 0, stream>>>(qh, kh, vth, out);
}
